// BiMamba2_4827543241054
// MI455X (gfx1250) — compile-verified, matrix operand order checked by reading, not run on hardware
//
#include <hip/hip_runtime.h>

// ---------------------------------------------------------------------------
// BiMamba2 fused pipeline for MI455X (gfx1250, wave32, WMMA bf16).
//
//  1. cvt fp32->bf16: x, in_proj_w, fuse_w halves; transpose ssm_out_w.
//  2. GEMM (WMMA): zxbcdt_bf16 = x @ in_proj_w^T        (16384 x 2320, K=512)
//     -- shared by fwd and rev directions (in_proj commutes with time flip).
//  3. GEMM (WMMA) x2: Wf/Wr = fuse_half @ ssm_out_w -> packed combined
//     weight Wcomb (512 x 2048) so out-proj + fuse collapse to one GEMM.
//  4. conv+silu+softplus per direction (depthwise, D_CONV=4, causal).
//  5. SSD chunk scan: 1 WG per (batch, dir, head); 64x128 state in LDS
//     (fp32 master + bf16 mirror), raw B/C tiles DMA'd via
//     GLOBAL_LOAD_ASYNC_TO_LDS_B128 (overlapped with a wave32 shuffle scan
//     of the dt cumsum), 4 WMMA matmuls per chunk.
//  6. gate (silu(z)) + RMSNorm -> Ycat bf16 (16384 x 2048).
//  7. GEMM (WMMA): out = Ycat @ Wcomb^T + fuse_b        (16384 x 512, K=2048)
// ---------------------------------------------------------------------------

#define BATCH   2
#define SEQLEN  8192
#define DMODEL  512
#define DSTATE  128
#define DINNER  1024
#define NHEADS  16
#define HEADDIM 64
#define CONVDIM 1280
#define DINPROJ 2320
#define CHUNK   64
#define NCHUNK  (SEQLEN / CHUNK)

#if defined(__gfx1250__) && __has_builtin(__builtin_amdgcn_global_load_async_to_lds_b128) && __has_builtin(__builtin_amdgcn_s_wait_asynccnt)
#define HAVE_ASYNC 1
#else
#define HAVE_ASYNC 0
#endif

typedef unsigned short u16;
typedef __attribute__((ext_vector_type(16))) __bf16 v16bf;
typedef __attribute__((ext_vector_type(8)))  float  v8f;
typedef __attribute__((__vector_size__(4 * sizeof(int)))) int v4i;

union FragB {
    v16bf v;
    u16   h[16];
    uint4 q[2];
};

// Native bf16 convert (RNE).
__device__ inline u16 f2bf(float f) {
    return __builtin_bit_cast(u16, (__bf16)f);
}
__device__ inline float bf2f(u16 h) {
    unsigned u = ((unsigned)h) << 16;
    return __builtin_bit_cast(float, u);
}
__device__ inline v8f wmma_bf16(v16bf a, v16bf b, v8f c) {
    return __builtin_amdgcn_wmma_f32_16x16x32_bf16(false, a, false, b,
                                                   (short)0, c, false, false);
}

// A/B fragment for 16x16x32 bf16 WMMA from a row-major bf16 matrix.
// Lane L<16 holds row (row0+L), K {k0..k0+7, k0+16..k0+23};
// lane L>=16 holds row (row0+L-16), K {k0+8..k0+15, k0+24..k0+31}.
__device__ inline v16bf ld_frag(const u16* base, int ld, int row0, int k0, int lane) {
    FragB f;
    int r  = row0 + (lane & 15);
    int kl = k0 + ((lane >> 4) << 3);
    const u16* p = base + (size_t)r * ld + kl;
    f.q[0] = *(const uint4*)(p);
    f.q[1] = *(const uint4*)(p + 16);
    return f.v;
}
__device__ inline v16bf ld_frag_clamp(const u16* base, int ld, int row0, int k0,
                                      int lane, int rmax) {
    FragB f;
    int r = row0 + (lane & 15);
    if (r >= rmax) r = rmax - 1;
    int kl = k0 + ((lane >> 4) << 3);
    const u16* p = base + (size_t)r * ld + kl;
    f.q[0] = *(const uint4*)(p);
    f.q[1] = *(const uint4*)(p + 16);
    return f.v;
}

// ---------------------------------------------------------------------------
// Generic GEMM: C[M,N] = A[M,K] @ B[N,K]^T (+ bias[n]), bf16 in, fp32 acc.
// Block = 256 threads = 8 waves arranged 4x2; wave computes a 32x64 tile
// (2x4 WMMA -> 8 wmma per 6 fragment loads). Block tile 128x128.
// Requires M % 128 == 0 and K % 32 == 0; N edge handled by clamp+guard.
// ---------------------------------------------------------------------------
template <typename CT>
__global__ void __launch_bounds__(256)
k_gemm(const u16* __restrict__ A, int lda, const u16* __restrict__ B, int ldb,
       CT* __restrict__ C, int ldc, int M, int N, int K,
       const float* __restrict__ bias) {
    int lane = threadIdx.x & 31;
    int wv   = threadIdx.x >> 5;
    int m0   = blockIdx.y * 128 + (wv >> 1) * 32;
    int n0   = blockIdx.x * 128 + (wv & 1) * 64;

    v8f acc[2][4] = {};
    for (int k0 = 0; k0 < K; k0 += 32) {
        v16bf a0 = ld_frag(A, lda, m0, k0, lane);
        v16bf a1 = ld_frag(A, lda, m0 + 16, k0, lane);
        v16bf b0 = ld_frag_clamp(B, ldb, n0, k0, lane, N);
        v16bf b1 = ld_frag_clamp(B, ldb, n0 + 16, k0, lane, N);
        v16bf b2 = ld_frag_clamp(B, ldb, n0 + 32, k0, lane, N);
        v16bf b3 = ld_frag_clamp(B, ldb, n0 + 48, k0, lane, N);
        acc[0][0] = wmma_bf16(a0, b0, acc[0][0]);
        acc[0][1] = wmma_bf16(a0, b1, acc[0][1]);
        acc[0][2] = wmma_bf16(a0, b2, acc[0][2]);
        acc[0][3] = wmma_bf16(a0, b3, acc[0][3]);
        acc[1][0] = wmma_bf16(a1, b0, acc[1][0]);
        acc[1][1] = wmma_bf16(a1, b1, acc[1][1]);
        acc[1][2] = wmma_bf16(a1, b2, acc[1][2]);
        acc[1][3] = wmma_bf16(a1, b3, acc[1][3]);
    }
#pragma unroll
    for (int i = 0; i < 2; ++i)
#pragma unroll
        for (int j = 0; j < 4; ++j)
#pragma unroll
            for (int e = 0; e < 8; ++e) {
                int m = m0 + i * 16 + e + ((lane >> 4) << 3);
                int n = n0 + j * 16 + (lane & 15);
                if (n < N) {
                    float v = acc[i][j][e];
                    if (bias) v += bias[n];
                    if constexpr (sizeof(CT) == 2)
                        C[(size_t)m * ldc + n] = (CT)f2bf(v);
                    else
                        C[(size_t)m * ldc + n] = v;
                }
            }
}

// ---------------------------------------------------------------------------
// fp32 -> bf16 copy with src/dst column windows (also used to pack Wcomb).
// ---------------------------------------------------------------------------
__global__ void __launch_bounds__(256)
k_cvt(const float* __restrict__ src, int sld, int scol, u16* __restrict__ dst,
      int dld, int dcol, int rows, int cols) {
    int idx = blockIdx.x * 256 + threadIdx.x;
    if (idx >= rows * cols) return;
    int r = idx / cols, c = idx - r * cols;
    dst[(size_t)r * dld + dcol + c] = f2bf(src[(size_t)r * sld + scol + c]);
}

// fp32 [rows,cols] -> bf16 transposed [cols,rows]
__global__ void __launch_bounds__(256)
k_tr(const float* __restrict__ src, int rows, int cols, u16* __restrict__ dst) {
    int idx = blockIdx.x * 256 + threadIdx.x;
    if (idx >= rows * cols) return;
    int r = idx / cols, c = idx - r * cols;
    dst[(size_t)c * rows + r] = f2bf(src[(size_t)r * cols + c]);
}

// ---------------------------------------------------------------------------
// Depthwise causal conv (D_CONV=4) + SiLU + dt softplus, per direction.
// grid = (SEQLEN, BATCH, 2 dirs), block = 256.
// ---------------------------------------------------------------------------
__global__ void __launch_bounds__(256)
k_conv(const u16* __restrict__ zx, const float* __restrict__ conv_w,
       const float* __restrict__ conv_b, const float* __restrict__ dt_bias,
       u16* __restrict__ xbc, float* __restrict__ dts) {
    int t = blockIdx.x, b = blockIdx.y, d = blockIdx.z;
    size_t obase = (((size_t)d * BATCH + b) * SEQLEN + t);
    for (int c = threadIdx.x; c < CONVDIM; c += 256) {
        float acc = conv_b[c];
#pragma unroll
        for (int j = 0; j < 4; ++j) {
            int ti = t - 3 + j;
            if (ti >= 0) {
                int to = d ? (SEQLEN - 1 - ti) : ti;
                size_t zr = ((size_t)b * SEQLEN + to) * DINPROJ;
                acc += conv_w[c * 4 + j] * bf2f(zx[zr + DINNER + c]);
            }
        }
        acc = acc / (1.f + expf(-acc)); // SiLU
        xbc[obase * CONVDIM + c] = f2bf(acc);
    }
    if (threadIdx.x < NHEADS) {
        int to = d ? (SEQLEN - 1 - t) : t;
        size_t zr = ((size_t)b * SEQLEN + to) * DINPROJ;
        float v = bf2f(zx[zr + 2 * DINNER + 2 * DSTATE + threadIdx.x]) + dt_bias[threadIdx.x];
        float sp = (v > 20.f) ? v : log1pf(expf(v));
        dts[obase * NHEADS + threadIdx.x] = sp;
    }
}

// ---------------------------------------------------------------------------
// SSD chunk scan. grid.x = BATCH*2*NHEADS = 64 workgroups, 256 thr (8 waves).
// State: fp32 master S[64][128] + bf16 mirror Sbf (for WMMA B-fragments).
// Raw B/C tiles arrive via async global->LDS DMA, overlapped with the dt
// shuffle-scan; per-row decay factors precomputed once; raw x tile kept in
// LDS so the D-skip epilogue never re-touches global memory.
// Per chunk:  G = C@B^T  ->  M = tril(G)*exp(dAcum)  ->  Y = M@Xd + Cd@Sbf^T
//             S = exp(Aend)*S + Xd^T@Bd ; Sbf = bf16(S)
// ---------------------------------------------------------------------------
__global__ void __launch_bounds__(256)
k_ssd(const u16* __restrict__ xbc, const float* __restrict__ dts,
      const float* __restrict__ A_log, const float* __restrict__ D_skip,
      u16* __restrict__ ybf) {
    int id = blockIdx.x;
    int h  = id & 15;
    int d  = (id >> 4) & 1;
    int b  = id >> 5;

    __shared__ __align__(16) float S[HEADDIM][DSTATE];   // 32 KB fp32 master
    __shared__ __align__(16) u16 Sbf[HEADDIM][DSTATE];   // 16 KB bf16 mirror
    __shared__ __align__(16) u16 Braw[CHUNK][DSTATE];    // 16 KB (async DMA dst)
    __shared__ __align__(16) u16 Craw[CHUNK][DSTATE];    // 16 KB (async DMA dst)
    __shared__ __align__(16) u16 Cd[CHUNK][DSTATE];      // 16 KB  C*exp(Acum)
    __shared__ __align__(16) u16 BdT[DSTATE][CHUNK];     // 16 KB  (B*exp(Aend-Acum))^T
    __shared__ __align__(16) u16 XdT[HEADDIM][CHUNK];    // 8 KB   (x*dt)^T  [p][l]
    __shared__ __align__(16) u16 Xraw[CHUNK][HEADDIM];   // 8 KB   raw x  [l][p]
    __shared__ __align__(16) u16 Mm[CHUNK][CHUNK];       // 8 KB   masked G
    __shared__ float Acum[CHUNK];
    __shared__ float dtc[CHUNK];
    __shared__ float eDecC[CHUNK];   // exp(Acum[l])
    __shared__ float eDecF[CHUNK];   // exp(Aend - Acum[l])
    __shared__ float stot[2];

    const float Aneg = -expf(A_log[h]);
    const float Dh   = D_skip[h];
    size_t xbase = (((size_t)d * BATCH + b) * SEQLEN) * (size_t)CONVDIM;
    size_t dbase = (((size_t)d * BATCH + b) * SEQLEN) * (size_t)NHEADS;
    size_t ybase = (((size_t)d * BATCH + b) * SEQLEN) * (size_t)DINNER;

    int lane = threadIdx.x & 31;
    int wv   = threadIdx.x >> 5;

    for (int i = threadIdx.x; i < HEADDIM * DSTATE; i += 256) {
        ((float*)S)[i] = 0.f;
        ((u16*)Sbf)[i] = (u16)0;
    }
    __syncthreads();

    for (int ck = 0; ck < NCHUNK; ++ck) {
        int tbase = ck * CHUNK;

        // --- (a) kick off async DMA of raw B,C chunk tiles into LDS ---
#if HAVE_ASYNC
        {
            int l0 = threadIdx.x >> 4;              // 0..15
            int n8 = (threadIdx.x & 15) * 8;        // 0..120 step 8 (16B chunks)
#pragma unroll
            for (int it = 0; it < 4; ++it) {
                int ll = l0 + it * 16;
                const u16* gb = &xbc[xbase + (size_t)(tbase + ll) * CONVDIM + DINNER + n8];
                __builtin_amdgcn_global_load_async_to_lds_b128(
                    (v4i*)gb, (v4i*)&Braw[ll][n8], 0, 0);
                __builtin_amdgcn_global_load_async_to_lds_b128(
                    (v4i*)(gb + DSTATE), (v4i*)&Craw[ll][n8], 0, 0);
            }
        }
#endif

        // --- (b) dt load + wave32 shuffle scan of cumsum(A*dt), overlapped ---
        float dv = 0.f, sv = 0.f;
        if (threadIdx.x < CHUNK) {
            dv = dts[dbase + (size_t)(tbase + threadIdx.x) * NHEADS + h];
            dtc[threadIdx.x] = dv;
            sv = Aneg * dv;
#pragma unroll
            for (int off = 1; off < 32; off <<= 1) {
                float nb = __shfl_up(sv, off, 32);
                if (lane >= off) sv += nb;
            }
            if (lane == 31) stot[wv] = sv;
        }
        __syncthreads();
        if (threadIdx.x >= 32 && threadIdx.x < CHUNK) sv += stot[0];
        if (threadIdx.x < CHUNK) Acum[threadIdx.x] = sv;
        __syncthreads();
        float Aend = Acum[CHUNK - 1];
        if (threadIdx.x < CHUNK) {
            eDecC[threadIdx.x] = expf(sv);
            eDecF[threadIdx.x] = expf(Aend - sv);
        }
#if HAVE_ASYNC
        __builtin_amdgcn_s_wait_asynccnt(0);
#endif
        __syncthreads();

        // --- (c) transform staging: decayed B^T, decayed C, Xraw, Xd^T ---
#if HAVE_ASYNC
        for (int i = threadIdx.x; i < CHUNK * DSTATE; i += 256) {
            int l = i >> 7, n = i & (DSTATE - 1);
            BdT[n][l] = f2bf(bf2f(Braw[l][n]) * eDecF[l]);
            Cd[l][n]  = f2bf(bf2f(Craw[l][n]) * eDecC[l]);
        }
#else
        for (int i = threadIdx.x; i < CHUNK * DSTATE; i += 256) {
            int l = i >> 7, n = i & (DSTATE - 1);
            size_t row = xbase + (size_t)(tbase + l) * CONVDIM;
            u16 bv = xbc[row + DINNER + n];
            u16 cv = xbc[row + DINNER + DSTATE + n];
            Braw[l][n] = bv;
            Craw[l][n] = cv;
            BdT[n][l]  = f2bf(bf2f(bv) * eDecF[l]);
            Cd[l][n]   = f2bf(bf2f(cv) * eDecC[l]);
        }
#endif
        for (int i = threadIdx.x; i < CHUNK * HEADDIM; i += 256) {
            int l = i >> 6, p = i & (HEADDIM - 1);
            u16 xv = xbc[xbase + (size_t)(tbase + l) * CONVDIM + h * HEADDIM + p];
            Xraw[l][p] = xv;
            XdT[p][l]  = f2bf(bf2f(xv) * dtc[l]);
        }
        __syncthreads();

        // --- matmul1: G = Craw @ Braw^T (64x64, K=128); mask+decay -> Mm ---
        int mt = wv >> 1;
#pragma unroll
        for (int jj = 0; jj < 2; ++jj) {
            int nt = ((wv & 1) << 1) + jj;
            v8f g = {};
#pragma unroll
            for (int kk = 0; kk < 4; ++kk) {
                v16bf a  = ld_frag(&Craw[0][0], DSTATE, mt * 16, kk * 32, lane);
                v16bf bb = ld_frag(&Braw[0][0], DSTATE, nt * 16, kk * 32, lane);
                g = wmma_bf16(a, bb, g);
            }
#pragma unroll
            for (int e = 0; e < 8; ++e) {
                int l = mt * 16 + e + ((lane >> 4) << 3);
                int s = nt * 16 + (lane & 15);
                float val = (l >= s) ? g[e] * expf(Acum[l] - Acum[s]) : 0.f;
                Mm[l][s] = f2bf(val);
            }
        }
        __syncthreads();

        // --- matmul2+3: Y = Mm @ XdT^T + Cd @ Sbf^T ; add D*x ; store ---
#pragma unroll
        for (int jj = 0; jj < 2; ++jj) {
            int nt = ((wv & 1) << 1) + jj;
            v8f y = {};
#pragma unroll
            for (int kk = 0; kk < 2; ++kk) {
                v16bf a  = ld_frag(&Mm[0][0], CHUNK, mt * 16, kk * 32, lane);
                v16bf bb = ld_frag(&XdT[0][0], CHUNK, nt * 16, kk * 32, lane);
                y = wmma_bf16(a, bb, y);
            }
#pragma unroll
            for (int kk = 0; kk < 4; ++kk) {
                v16bf a  = ld_frag(&Cd[0][0], DSTATE, mt * 16, kk * 32, lane);
                v16bf bb = ld_frag(&Sbf[0][0], DSTATE, nt * 16, kk * 32, lane);
                y = wmma_bf16(a, bb, y);
            }
#pragma unroll
            for (int e = 0; e < 8; ++e) {
                int l = mt * 16 + e + ((lane >> 4) << 3);
                int p = nt * 16 + (lane & 15);
                int tt = tbase + l;
                int to = d ? (SEQLEN - 1 - tt) : tt;
                float xv = bf2f(Xraw[l][p]);
                ybf[ybase + (size_t)to * DINNER + h * HEADDIM + p] = f2bf(y[e] + Dh * xv);
            }
        }
        __syncthreads();

        // --- matmul4: S = exp(Aend)*S + XdT @ BdT^T (64x128, K=64) ---
        float eA = expf(Aend);
        int pt = wv >> 1;
#pragma unroll
        for (int jj = 0; jj < 4; ++jj) {
            int nt = ((wv & 1) << 2) + jj;
            v8f sa;
#pragma unroll
            for (int e = 0; e < 8; ++e)
                sa[e] = eA * S[pt * 16 + e + ((lane >> 4) << 3)][nt * 16 + (lane & 15)];
#pragma unroll
            for (int kk = 0; kk < 2; ++kk) {
                v16bf a  = ld_frag(&XdT[0][0], CHUNK, pt * 16, kk * 32, lane);
                v16bf bb = ld_frag(&BdT[0][0], CHUNK, nt * 16, kk * 32, lane);
                sa = wmma_bf16(a, bb, sa);
            }
#pragma unroll
            for (int e = 0; e < 8; ++e) {
                int p = pt * 16 + e + ((lane >> 4) << 3);
                int n = nt * 16 + (lane & 15);
                S[p][n]   = sa[e];
                Sbf[p][n] = f2bf(sa[e]);
            }
        }
        __syncthreads();
    }
}

// ---------------------------------------------------------------------------
// Gate (y * silu(z)) + RMSNorm over 1024 -> Ycat bf16 [16384][2048].
// grid.x = BATCH*SEQLEN blocks, 256 threads (4 elements/thread/dir).
// ---------------------------------------------------------------------------
__global__ void __launch_bounds__(256)
k_gate(const u16* __restrict__ ybf, const u16* __restrict__ zx,
       const float* __restrict__ norm_w, u16* __restrict__ ycat) {
    int bt = blockIdx.x;
    __shared__ float red[8];
    int lane = threadIdx.x & 31, wv = threadIdx.x >> 5;

    for (int d = 0; d < 2; ++d) {
        float vals[4];
        float ss = 0.f;
#pragma unroll
        for (int q = 0; q < 4; ++q) {
            int i = threadIdx.x + q * 256;
            float z = bf2f(zx[(size_t)bt * DINPROJ + i]);
            float y = bf2f(ybf[((size_t)d * BATCH * SEQLEN + bt) * DINNER + i]);
            float g = y * (z / (1.f + expf(-z)));
            vals[q] = g;
            ss += g * g;
        }
#pragma unroll
        for (int off = 16; off >= 1; off >>= 1) ss += __shfl_xor(ss, off, 32);
        if (lane == 0) red[wv] = ss;
        __syncthreads();
        float tot = 0.f;
#pragma unroll
        for (int w = 0; w < 8; ++w) tot += red[w];
        float rinv = rsqrtf(tot / (float)DINNER + 1e-5f);
#pragma unroll
        for (int q = 0; q < 4; ++q) {
            int i = threadIdx.x + q * 256;
            ycat[(size_t)bt * (2 * DINNER) + d * DINNER + i] = f2bf(vals[q] * rinv * norm_w[i]);
        }
        __syncthreads();
    }
}

// ---------------------------------------------------------------------------
extern "C" void kernel_launch(void* const* d_in, const int* in_sizes, int n_in,
                              void* d_out, int out_size, void* d_ws, size_t ws_size,
                              hipStream_t stream) {
    const float* x         = (const float*)d_in[0];
    const float* in_proj_w = (const float*)d_in[1];
    const float* conv_w    = (const float*)d_in[2];
    const float* conv_b    = (const float*)d_in[3];
    const float* dt_bias   = (const float*)d_in[4];
    const float* A_log     = (const float*)d_in[5];
    const float* D_skip    = (const float*)d_in[6];
    const float* norm_w    = (const float*)d_in[7];
    const float* ssm_out_w = (const float*)d_in[8];
    const float* fuse_w    = (const float*)d_in[9];
    const float* fuse_b    = (const float*)d_in[10];
    float* out = (float*)d_out;

    const size_t M = (size_t)BATCH * SEQLEN; // 16384

    // ---- workspace partition (~325 MB) ----
    char* wp = (char*)d_ws;
    auto alloc = [&](size_t bytes) {
        void* p = (void*)wp;
        wp += (bytes + 255) & ~(size_t)255;
        return p;
    };
    u16*   xbf   = (u16*)alloc(M * DMODEL * 2);                 // x bf16
    u16*   wbf   = (u16*)alloc((size_t)DINPROJ * DMODEL * 2);   // in_proj_w bf16
    u16*   zxbf  = (u16*)alloc(M * DINPROJ * 2);                // zxbcdt bf16
    u16*   xbc   = (u16*)alloc((size_t)2 * M * CONVDIM * 2);    // conv act, per dir
    float* dts   = (float*)alloc((size_t)2 * M * NHEADS * 4);   // softplus dt
    u16*   ybf   = (u16*)alloc((size_t)2 * M * DINNER * 2);     // ssd out, per dir
    u16*   ycat  = (u16*)alloc(M * 2 * DINNER * 2);             // gated/normed concat
    u16*   sowT  = (u16*)alloc((size_t)DINNER * DMODEL * 2);    // ssm_out_w^T bf16
    u16*   fh0   = (u16*)alloc((size_t)DMODEL * DMODEL * 2);    // fuse_w[:, :512]
    u16*   fh1   = (u16*)alloc((size_t)DMODEL * DMODEL * 2);    // fuse_w[:, 512:]
    float* wtmp  = (float*)alloc((size_t)DMODEL * DINNER * 4);  // Wf/Wr fp32 temp
    u16*   wcomb = (u16*)alloc((size_t)DMODEL * 2 * DINNER * 2);// combined 512x2048

    auto g1 = [](size_t n) { return dim3((unsigned)((n + 255) / 256)); };

    // 1) conversions
    k_cvt<<<g1(M * DMODEL), 256, 0, stream>>>(x, DMODEL, 0, xbf, DMODEL, 0, (int)M, DMODEL);
    k_cvt<<<g1((size_t)DINPROJ * DMODEL), 256, 0, stream>>>(in_proj_w, DMODEL, 0, wbf, DMODEL, 0, DINPROJ, DMODEL);
    k_tr<<<g1((size_t)DMODEL * DINNER), 256, 0, stream>>>(ssm_out_w, DMODEL, DINNER, sowT);
    k_cvt<<<g1((size_t)DMODEL * DMODEL), 256, 0, stream>>>(fuse_w, 2 * DMODEL, 0, fh0, DMODEL, 0, DMODEL, DMODEL);
    k_cvt<<<g1((size_t)DMODEL * DMODEL), 256, 0, stream>>>(fuse_w, 2 * DMODEL, DMODEL, fh1, DMODEL, 0, DMODEL, DMODEL);

    // 2) in_proj GEMM (shared by both directions): zx = x @ in_proj_w^T
    k_gemm<u16><<<dim3((DINPROJ + 127) / 128, (unsigned)(M / 128)), 256, 0, stream>>>(
        xbf, DMODEL, wbf, DMODEL, zxbf, DINPROJ, (int)M, DINPROJ, DMODEL, nullptr);

    // 3) combined output weight: Wcomb = [fuse_half0 @ ssm_out_w | fuse_half1 @ ssm_out_w]
    k_gemm<float><<<dim3(DINNER / 128, DMODEL / 128), 256, 0, stream>>>(
        fh0, DMODEL, sowT, DMODEL, wtmp, DINNER, DMODEL, DINNER, DMODEL, nullptr);
    k_cvt<<<g1((size_t)DMODEL * DINNER), 256, 0, stream>>>(wtmp, DINNER, 0, wcomb, 2 * DINNER, 0, DMODEL, DINNER);
    k_gemm<float><<<dim3(DINNER / 128, DMODEL / 128), 256, 0, stream>>>(
        fh1, DMODEL, sowT, DMODEL, wtmp, DINNER, DMODEL, DINNER, DMODEL, nullptr);
    k_cvt<<<g1((size_t)DMODEL * DINNER), 256, 0, stream>>>(wtmp, DINNER, 0, wcomb, 2 * DINNER, DINNER, DMODEL, DINNER);

    // 4) depthwise conv + silu + softplus(dt), both directions
    k_conv<<<dim3(SEQLEN, BATCH, 2), 256, 0, stream>>>(zxbf, conv_w, conv_b, dt_bias, xbc, dts);

    // 5) SSD chunk scan, 1 WG per (b, dir, head)
    k_ssd<<<dim3(BATCH * 2 * NHEADS), 256, 0, stream>>>(xbc, dts, A_log, D_skip, ybf);

    // 6) gate + RMSNorm -> Ycat
    k_gate<<<dim3((unsigned)M), 256, 0, stream>>>(ybf, zxbf, norm_w, ycat);

    // 7) final fused GEMM: out = Ycat @ Wcomb^T + fuse_b
    k_gemm<float><<<dim3(DMODEL / 128, (unsigned)(M / 128)), 256, 0, stream>>>(
        ycat, 2 * DINNER, wcomb, 2 * DINNER, out, DMODEL, (int)M, DMODEL, 2 * DINNER, fuse_b);

    (void)in_sizes; (void)n_in; (void)out_size; (void)ws_size;
}
